// PytorchHMM_16801912062634
// MI455X (gfx1250) — matrix operand order repaired and checked by reading, not compiled, not run on hardware
//
#include <hip/hip_runtime.h>
#include <hip/hip_bf16.h>
#include <math.h>

#define HID   64
#define OBSC  4
#define BATCH 128
#define SEQ   2048
#define BTILE 16

#define LOG2E 1.4426950408889634f
#define LN2   0.6931471805599453f

typedef __attribute__((ext_vector_type(16))) __bf16 v16bf;
typedef __attribute__((ext_vector_type(8)))  float  v8f;
typedef __attribute__((ext_vector_type(4)))  float  v4f;

__device__ __forceinline__ unsigned short f32_to_bf16_rne(float x) {
  unsigned int u = __float_as_uint(x);
  u += 0x7fffu + ((u >> 16) & 1u);
  return (unsigned short)(u >> 16);
}

// ---------------------------------------------------------------------------
// Setup: softmax the transition rows (+eps) -> bf16 row-major [i][j],
// emission coefficients (em = c0[h] + sum_c w1*x + w2*x^2), log initial dist.
// ---------------------------------------------------------------------------
__global__ void hmm_setup(const float* __restrict__ trans_logits,
                          const float* __restrict__ init_logits,
                          const float* __restrict__ means,
                          const float* __restrict__ log_vars,
                          unsigned short* __restrict__ Tbf,   // [64][64] bf16
                          float* __restrict__ coef,           // [64][12]
                          float* __restrict__ log_init) {     // [64]
  int h = threadIdx.x;
  if (h >= HID) return;

  float m = -INFINITY;
  for (int j = 0; j < HID; ++j) m = fmaxf(m, trans_logits[h * HID + j]);
  float z = 0.f;
  for (int j = 0; j < HID; ++j)
    z += __builtin_amdgcn_exp2f((trans_logits[h * HID + j] - m) * LOG2E);
  float invz = 1.0f / z;
  for (int j = 0; j < HID; ++j) {
    float p = __builtin_amdgcn_exp2f((trans_logits[h * HID + j] - m) * LOG2E) * invz + 1e-10f;
    Tbf[h * HID + j] = f32_to_bf16_rne(p);
  }

  float c0 = 0.f;
  for (int c = 0; c < OBSC; ++c) {
    float v  = __builtin_amdgcn_exp2f(log_vars[h * OBSC + c] * LOG2E) + 1e-6f;
    float iv = 1.0f / v;
    float mu = means[h * OBSC + c];
    coef[h * 12 + 1 + c] = mu * iv;      // w1
    coef[h * 12 + 5 + c] = -0.5f * iv;   // w2
    c0 += LN2 * __builtin_amdgcn_logf(2.0f * 3.14159265358979f * v) + mu * mu * iv;
  }
  coef[h * 12 + 0] = -0.5f * c0;

  float mi = -INFINITY;
  for (int j = 0; j < HID; ++j) mi = fmaxf(mi, init_logits[j]);
  float zi = 0.f;
  for (int j = 0; j < HID; ++j)
    zi += __builtin_amdgcn_exp2f((init_logits[j] - mi) * LOG2E);
  float p0 = __builtin_amdgcn_exp2f((init_logits[h] - mi) * LOG2E) / zi;
  log_init[h] = LN2 * __builtin_amdgcn_logf(p0 + 1e-10f);
}

// ---------------------------------------------------------------------------
// Forward recursion. One workgroup = 16 batch rows; 4 waves, wave w owns
// output states j in [16w, 16w+16). B tiles of the transition matrix live in
// VGPRs for the whole 2048-step loop. 3 barriers per step.
// ---------------------------------------------------------------------------
__global__ __launch_bounds__(128)
void hmm_forward(const float* __restrict__ obs,            // [B,S,4]
                 const unsigned short* __restrict__ Tbf,   // [64][64] bf16
                 const float* __restrict__ coef,           // [64][12]
                 const float* __restrict__ log_init,       // [64]
                 float* __restrict__ alpha_out,            // [B,S,64]
                 float* __restrict__ ll_out) {             // [B]
  __shared__ __align__(16) float          s_alpha[BTILE][HID + 1];
  __shared__ __align__(16) unsigned short s_p[BTILE][HID];
  __shared__ float s_m[BTILE];
  __shared__ float s_obs[BTILE][OBSC];
  __shared__ float s_obs2[BTILE][OBSC];
  __shared__ float s_red[BTILE][8];

  const int tid  = threadIdx.x;
  const int lane = tid & 31;
  const int wave = tid >> 5;
  const int b0   = blockIdx.x * BTILE;

  const int nloc = lane & 15;        // column within 16x16 tile
  const int hi   = (lane >> 4) & 1;  // lane half (D rows +8, A/B K offset)
  const int j    = wave * 16 + nloc; // this lane's hidden state column

  // loop-invariant emission coefficients for state j
  const float c0 = coef[j * 12 + 0];
  float w1[OBSC], w2[OBSC];
#pragma unroll
  for (int c = 0; c < OBSC; ++c) {
    w1[c] = coef[j * 12 + 1 + c];
    w2[c] = coef[j * 12 + 5 + c];
  }

  // loop-invariant output base per D register (row r = v + 8*hi, col j)
  size_t obase[8];
#pragma unroll
  for (int v = 0; v < 8; ++v)
    obase[v] = (size_t)(b0 + v + hi * 8) * SEQ * HID + (size_t)j;

  // loop-invariant B tiles: B is 32x16 (K x N) bf16 per the ISA layout:
// VGPR v, lanes 0-15 hold K = {2v,2v+1} (+8 for v>=4), +8 for lanes 16-31, N = lane&15.
  union BU { unsigned short us[16]; v16bf v; };
  v16bf Bt0, Bt1;
  {
    BU bu0, bu1;
#pragma unroll
    for (int e = 0; e < 16; ++e) {
      int K0 = (hi ? 8 : 0) + (e < 8 ? e : e + 8);        // K chunk 0..31
      int K1 = 32 + (hi ? 8 : 0) + (e < 8 ? e : e + 8);   // K chunk 32..63
      bu0.us[e] = Tbf[K0 * HID + j];
      bu1.us[e] = Tbf[K1 * HID + j];
    }
    Bt0 = bu0.v;
    Bt1 = bu1.v;
  }

  const int rr  = tid & 15;  // reduction row
  const int seg = tid >> 4;  // reduction segment (0..7)

  // ---- t = 0: alpha0 = log_init[j] + em(obs[:,0,:]) ----
  if (tid < BTILE * OBSC) {
    int r = tid >> 2, c = tid & 3;
    float x = obs[((size_t)(b0 + r) * SEQ + 0) * OBSC + c];
    s_obs[r][c]  = x;
    s_obs2[r][c] = x * x;
  }
  __syncthreads();
#pragma unroll
  for (int e = 0; e < 8; ++e) {
    int idx = tid * 8 + e;
    int r = idx >> 6, jj = idx & 63;
    float em = coef[jj * 12 + 0];
#pragma unroll
    for (int c = 0; c < OBSC; ++c)
      em += coef[jj * 12 + 1 + c] * s_obs[r][c] + coef[jj * 12 + 5 + c] * s_obs2[r][c];
    s_alpha[r][jj] = log_init[jj] + em;
  }
  __syncthreads();
  {
    int r = tid >> 3, jo = (tid & 7) * 8;
    size_t base = ((size_t)(b0 + r) * SEQ + 0) * HID + jo;
    v4f x0 = {s_alpha[r][jo + 0], s_alpha[r][jo + 1], s_alpha[r][jo + 2], s_alpha[r][jo + 3]};
    v4f x1 = {s_alpha[r][jo + 4], s_alpha[r][jo + 5], s_alpha[r][jo + 6], s_alpha[r][jo + 7]};
    __builtin_nontemporal_store(x0, (v4f*)(alpha_out + base));
    __builtin_nontemporal_store(x1, (v4f*)(alpha_out + base + 4));
  }

  // ---- recursion over time ----
  for (int t = 1; t < SEQ; ++t) {
    // P1: partial row max (all threads) + fetch obs(t) (64 threads)
    float pm = -INFINITY;
#pragma unroll
    for (int k = 0; k < 8; ++k) pm = fmaxf(pm, s_alpha[rr][seg * 8 + k]);
    s_red[rr][seg] = pm;
    if (tid < BTILE * OBSC) {
      int r = tid >> 2, c = tid & 3;
      float x = obs[((size_t)(b0 + r) * SEQ + t) * OBSC + c];
      s_obs[r][c]  = x;
      s_obs2[r][c] = x * x;
    }
    __syncthreads();

    // P2: finish row max from broadcast partials; p = exp2((a-m)*log2e) -> bf16
    {
      float m = s_red[rr][0];
#pragma unroll
      for (int k = 1; k < 8; ++k) m = fmaxf(m, s_red[rr][k]);
      if (seg == 0) s_m[rr] = m;
#pragma unroll
      for (int k = 0; k < 8; ++k) {
        int jj = seg * 8 + k;
        float p = __builtin_amdgcn_exp2f((s_alpha[rr][jj] - m) * LOG2E);
        s_p[rr][jj] = f32_to_bf16_rne(p);
      }
    }
    __syncthreads();

    // P3: load A tiles (16x32 bf16, ISA layout), 2x WMMA, fuse log+em,
    //     write s_alpha (for next step) + direct NT store to global.
    union AU { unsigned int u[8]; v16bf v; };
    AU a0, a1;
    const int M = lane & 15;
#pragma unroll
    for (int v = 0; v < 8; ++v) {
      int kbase = 2 * v + (v >= 4 ? 8 : 0) + (hi ? 8 : 0);
      a0.u[v] = *(const unsigned int*)&s_p[M][kbase];
      a1.u[v] = *(const unsigned int*)&s_p[M][32 + kbase];
    }
    v8f acc = {};
    acc = __builtin_amdgcn_wmma_f32_16x16x32_bf16(false, a0.v, false, Bt0,
                                                  (short)0, acc, false, false);
    acc = __builtin_amdgcn_wmma_f32_16x16x32_bf16(false, a1.v, false, Bt1,
                                                  (short)0, acc, false, false);
#pragma unroll
    for (int v = 0; v < 8; ++v) {
      int r = v + hi * 8;                 // D layout: VGPR v -> row v (+8 hi)
      float em = c0;
#pragma unroll
      for (int c = 0; c < OBSC; ++c)
        em += w1[c] * s_obs[r][c] + w2[c] * s_obs2[r][c];
      float a = s_m[r] + LN2 * __builtin_amdgcn_logf(acc[v]) + em;
      s_alpha[r][j] = a;
      __builtin_nontemporal_store(a, alpha_out + obase[v] + (size_t)t * HID);
    }
    __syncthreads();
  }

  // ---- final log-likelihood ----
  if (tid < BTILE) {
    float m = -INFINITY;
    for (int jj = 0; jj < HID; ++jj) m = fmaxf(m, s_alpha[tid][jj]);
    float z = 0.f;
    for (int jj = 0; jj < HID; ++jj)
      z += __builtin_amdgcn_exp2f((s_alpha[tid][jj] - m) * LOG2E);
    ll_out[b0 + tid] = m + LN2 * __builtin_amdgcn_logf(z);
  }
}

// ---------------------------------------------------------------------------
extern "C" void kernel_launch(void* const* d_in, const int* in_sizes, int n_in,
                              void* d_out, int out_size, void* d_ws, size_t ws_size,
                              hipStream_t stream) {
  const float* obs = (const float*)d_in[0];  // [128,2048,4]
  const float* tl  = (const float*)d_in[1];  // [64,64]
  const float* il  = (const float*)d_in[2];  // [64]
  const float* mu  = (const float*)d_in[3];  // [64,4]
  const float* lv  = (const float*)d_in[4];  // [64,4]

  char* ws = (char*)d_ws;
  unsigned short* Tbf = (unsigned short*)ws;            // 8192 B
  float* coef         = (float*)(ws + 8192);            // 3072 B
  float* log_init     = (float*)(ws + 8192 + 3072);     //  256 B

  float* alpha_out = (float*)d_out;
  float* ll_out    = alpha_out + (size_t)BATCH * SEQ * HID;

  hmm_setup<<<1, 64, 0, stream>>>(tl, il, mu, lv, Tbf, coef, log_init);
  hmm_forward<<<BATCH / BTILE, 128, 0, stream>>>(obs, Tbf, coef, log_init,
                                                 alpha_out, ll_out);
}
